// LSTMLayer_82927228551933
// MI455X (gfx1250) — matrix operand order repaired and checked by reading, not compiled, not run on hardware
//
#include <hip/hip_runtime.h>

#define T_STEPS 512
#define BATCH   64
#define INDIM   512
#define HID     512
#define GATES   2048  // 4*HID
#define NTILES  (GATES / 16)   // 128 col tiles
#define MTILES_PER_STEP (BATCH / 16)  // 4 row tiles per timestep

typedef __attribute__((ext_vector_type(16))) __bf16 v16bf;
typedef __attribute__((ext_vector_type(8)))  __bf16 v8bf;
typedef __attribute__((ext_vector_type(8)))  float  v8f;

__device__ __forceinline__ float sigmoid_f(float x) {
  return 1.0f / (1.0f + __expf(-x));
}
__device__ __forceinline__ float tanh_f(float x) {
  float e = __expf(2.0f * x);
  return 1.0f - 2.0f / (e + 1.0f);
}

__global__ __launch_bounds__(256) void f32_to_bf16_kernel(
    const float* __restrict__ src, __bf16* __restrict__ dst, int n) {
  int i = blockIdx.x * blockDim.x + threadIdx.x;
  if (i < n) dst[i] = (__bf16)src[i];
}

// x_gates = x_bf @ w_ih_bf^T + b_ih + b_hh, fragment-major tile layout.
__global__ __launch_bounds__(256) void xgemm_kernel(
    const __bf16* __restrict__ xb,    // [M, INDIM]
    const __bf16* __restrict__ wb,    // [GATES, INDIM]
    const float*  __restrict__ b_ih,  // [GATES]
    const float*  __restrict__ b_hh,  // [GATES]
    float* __restrict__ xg)           // [M/16][NTILES][256] fragment-major
{
  const int M = T_STEPS * BATCH;
  const int tilesM = M / 64;
  int bx    = blockIdx.x;
  int tileM = bx % tilesM;
  int tileN = bx / tilesM;
  int wid   = threadIdx.x >> 5;
  int lane  = threadIdx.x & 31;
  int waveM = wid & 3;
  int waveN = wid >> 2;
  int m_base = tileM * 64 + waveM * 16;
  int n_base = tileN * 128 + waveN * 64;
  int lrow = lane & 15;
  int lgrp = lane >> 4;

  v8f acc[4] = {};
  const __bf16* arow = xb + (size_t)(m_base + lrow) * INDIM;

  for (int kk = 0; kk < INDIM; kk += 32) {
    v8bf alo = __builtin_nontemporal_load((const v8bf*)(arow + kk + lgrp * 16));
    v8bf ahi = __builtin_nontemporal_load((const v8bf*)(arow + kk + 8 + lgrp * 16));
    v16bf a;
#pragma unroll
    for (int e = 0; e < 8; ++e) { a[e] = alo[e]; a[8 + e] = ahi[e]; }

#pragma unroll
    for (int j = 0; j < 4; ++j) {
      int n = n_base + 16 * j + lrow;
      v16bf b = *(const v16bf*)(wb + (size_t)n * INDIM + kk + lgrp * 16);
      acc[j] = __builtin_amdgcn_wmma_f32_16x16x32_bf16(
          false, a, false, b, (short)0, acc[j], false, false);
    }
  }

  int row_tile = m_base >> 4;
#pragma unroll
  for (int j = 0; j < 4; ++j) {
    int n = n_base + 16 * j + lrow;
    float bias = b_ih[n] + b_hh[n];
    int col_tile = (n_base >> 4) + j;
    float* tile = xg + ((size_t)row_tile * NTILES + col_tile) * 256 + lane * 8;
    *(v8f*)tile = acc[j] + bias;
  }
}

// One fused LSTM timestep; two independent WMMA accumulator chains.
__global__ __launch_bounds__(128) void lstm_step_kernel(
    const __bf16* __restrict__ h_prev_bf,
    const float*  __restrict__ c_prev,
    const __bf16* __restrict__ whb,
    const float*  __restrict__ xg_t,
    float*  __restrict__ h_out,
    __bf16* __restrict__ h_next_bf,
    float*  __restrict__ c_out,
    float*  __restrict__ hT_out,
    float*  __restrict__ cT_out)
{
  __shared__ float gsm[4][16][16];

  int wg     = blockIdx.x;
  int m_tile = wg & 3;
  int h_tile = wg >> 2;
  int wid    = threadIdx.x >> 5;
  int lane   = threadIdx.x & 31;
  int lrow   = lane & 15;
  int lgrp   = lane >> 4;
  int m_base = m_tile * 16;
  int n_base = wid * HID + h_tile * 16;

  v8f acc0 = {}, acc1 = {};
  const __bf16* arow = h_prev_bf + (size_t)(m_base + lrow) * HID;
  const __bf16* brow = whb + (size_t)(n_base + lrow) * HID;

  for (int kk = 0; kk < HID; kk += 64) {
    __builtin_prefetch(brow + kk + 128, 0, 0);

    v8bf alo0 = *(const v8bf*)(arow + kk + lgrp * 16);
    v8bf ahi0 = *(const v8bf*)(arow + kk + 8 + lgrp * 16);
    v16bf a0;
#pragma unroll
    for (int e = 0; e < 8; ++e) { a0[e] = alo0[e]; a0[8 + e] = ahi0[e]; }
    v16bf b0 = *(const v16bf*)(brow + kk + lgrp * 16);
    acc0 = __builtin_amdgcn_wmma_f32_16x16x32_bf16(
        false, a0, false, b0, (short)0, acc0, false, false);

    v8bf alo1 = *(const v8bf*)(arow + kk + 32 + lgrp * 16);
    v8bf ahi1 = *(const v8bf*)(arow + kk + 40 + lgrp * 16);
    v16bf a1;
#pragma unroll
    for (int e = 0; e < 8; ++e) { a1[e] = alo1[e]; a1[8 + e] = ahi1[e]; }
    v16bf b1 = *(const v16bf*)(brow + kk + 32 + lgrp * 16);
    acc1 = __builtin_amdgcn_wmma_f32_16x16x32_bf16(
        false, a1, false, b1, (short)0, acc1, false, false);
  }

  int col_tile = wid * 32 + h_tile;
  const float* xtile =
      xg_t + ((size_t)m_tile * NTILES + col_tile) * 256 + lane * 8;
  v8f xv = *(const v8f*)xtile;
  v8f acc = acc0 + acc1 + xv;

#pragma unroll
  for (int r = 0; r < 8; ++r) {
    gsm[wid][r + 8 * lgrp][lrow] = acc[r];
  }
  __syncthreads();

  for (int e = threadIdx.x; e < 256; e += 128) {
    int mi = e >> 4, hi = e & 15;
    int b  = m_base + mi;
    int h  = h_tile * 16 + hi;
    float ig = sigmoid_f(gsm[0][mi][hi]);
    float fg = sigmoid_f(gsm[1][mi][hi]);
    float gg = tanh_f  (gsm[2][mi][hi]);
    float og = sigmoid_f(gsm[3][mi][hi]);
    size_t idx = (size_t)b * HID + h;
    float cp = c_prev[idx];
    float cn = fg * cp + ig * gg;
    float hn = og * tanh_f(cn);
    c_out[idx] = cn;
    h_out[idx] = hn;
    h_next_bf[idx] = (__bf16)hn;
    if (hT_out) { hT_out[idx] = hn; cT_out[idx] = cn; }
  }
}

extern "C" void kernel_launch(void* const* d_in, const int* in_sizes, int n_in,
                              void* d_out, int out_size, void* d_ws, size_t ws_size,
                              hipStream_t stream) {
  (void)in_sizes; (void)n_in; (void)out_size; (void)ws_size;

  const float* x   = (const float*)d_in[0];
  const float* h0  = (const float*)d_in[1];
  const float* c0  = (const float*)d_in[2];
  const float* wih = (const float*)d_in[3];
  const float* whh = (const float*)d_in[4];
  const float* bih = (const float*)d_in[5];
  const float* bhh = (const float*)d_in[6];
  float* out = (float*)d_out;

  char* ws = (char*)d_ws;
  size_t off = 0;
  __bf16* wih_bf = (__bf16*)(ws + off); off += (size_t)GATES * INDIM * 2;
  __bf16* whh_bf = (__bf16*)(ws + off); off += (size_t)GATES * HID * 2;
  __bf16* x_bf   = (__bf16*)(ws + off); off += (size_t)T_STEPS * BATCH * INDIM * 2;
  float*  xg     = (float*) (ws + off); off += (size_t)T_STEPS * BATCH * GATES * 4;
  float*  c_buf  = (float*) (ws + off); off += (size_t)BATCH * HID * 4;
  __bf16* h_bf0  = (__bf16*)(ws + off); off += (size_t)BATCH * HID * 2;
  __bf16* h_bf1  = (__bf16*)(ws + off); off += (size_t)BATCH * HID * 2;

  int nw = GATES * INDIM;
  f32_to_bf16_kernel<<<(nw + 255) / 256, 256, 0, stream>>>(wih, wih_bf, nw);
  f32_to_bf16_kernel<<<(nw + 255) / 256, 256, 0, stream>>>(whh, whh_bf, nw);
  int nx = T_STEPS * BATCH * INDIM;
  f32_to_bf16_kernel<<<(nx + 255) / 256, 256, 0, stream>>>(x, x_bf, nx);
  int nh = BATCH * HID;
  f32_to_bf16_kernel<<<(nh + 255) / 256, 256, 0, stream>>>(h0, h_bf0, nh);

  const int Mrows = T_STEPS * BATCH;
  dim3 ggrid((Mrows / 64) * (GATES / 128));
  xgemm_kernel<<<ggrid, 256, 0, stream>>>(x_bf, wih_bf, bih, bhh, xg);

  float* outs = out;
  float* hT   = out + (size_t)T_STEPS * BATCH * HID;
  float* cT   = hT + (size_t)BATCH * HID;
  __bf16* hping[2] = { h_bf0, h_bf1 };

  const size_t xg_step = (size_t)MTILES_PER_STEP * NTILES * 256;

  for (int t = 0; t < T_STEPS; ++t) {
    const float* cp = (t == 0) ? c0 : c_buf;
    bool last = (t == T_STEPS - 1);
    lstm_step_kernel<<<128, 128, 0, stream>>>(
        hping[t & 1], cp, whh_bf, xg + (size_t)t * xg_step,
        outs + (size_t)t * BATCH * HID, hping[(t + 1) & 1], c_buf,
        last ? hT : nullptr, last ? cT : nullptr);
  }
}
